// DecoderBackbone_38697655337110
// MI455X (gfx1250) — compile-verified, matrix operand order checked by reading, not run on hardware
//
#include <hip/hip_runtime.h>
#include <hip/hip_bf16.h>

#define B_ 2
#define T_ 1024
#define HID_ 2048
#define NH_ 16
#define NKV_ 4
#define HD_ 128
#define INTER_ 5632
#define L_ 4
#define BT_ (B_ * T_)

typedef __attribute__((ext_vector_type(16))) __bf16 bf16x16;
typedef __attribute__((ext_vector_type(2)))  __bf16 bf16x2;
typedef __attribute__((ext_vector_type(8))) float floatx8;
typedef __attribute__((ext_vector_type(2))) float floatx2;
typedef __attribute__((ext_vector_type(4))) unsigned v4u;
typedef __attribute__((ext_vector_type(8))) int v8i_;
typedef __attribute__((ext_vector_type(4))) int v4i_;

// Native HW f32->bf16 convert (RNE).
__device__ __forceinline__ unsigned short f2bf(float f) {
  union { __bf16 b; unsigned short u; } c;
  c.b = (__bf16)f;
  return c.u;
}

// Packed pair convert: maps directly onto v_cvt_pk_bf16_f32 (no repacking).
__device__ __forceinline__ unsigned pack_bf2(float x, float y) {
  floatx2 f; f.x = x; f.y = y;
  union { bf16x2 v; unsigned u; } c;
  c.v = __builtin_convertvector(f, bf16x2);
  return c.u;
}

__device__ __forceinline__ floatx8 wmma_bf16(bf16x16 a, bf16x16 b, floatx8 c) {
  // (neg_a, A, neg_b, B, c_mod, C, reuse_a, reuse_b)
  return __builtin_amdgcn_wmma_f32_16x16x32_bf16(false, a, false, b, (short)0, c,
                                                 false, false);
}

// Load one 16x32 bf16 fragment (A-operand layout; B-operand when source is
// stored [N,K] row-major) from a row-major bf16 matrix with leading dim ld.
// Lane L (0-15): row L, K={0..7,16..23}; lane L+16: row L, K={8..15,24..31}.
__device__ __forceinline__ bf16x16 load_frag(const unsigned short* p0, int ld) {
  int lane = threadIdx.x & 31;
  int row = lane & 15, half = lane >> 4;
  union { bf16x16 v; unsigned u[8]; } f;
  const unsigned short* p = p0 + (size_t)row * ld + half * 8;
#pragma unroll
  for (int j = 0; j < 8; ++j) {
    int k0 = (j & 3) * 2 + (j >> 2) * 16;
    f.u[j] = *(const unsigned*)(p + k0);
  }
  return f.v;
}

__device__ __forceinline__ float hmax16(float v) {
#pragma unroll
  for (int m = 1; m < 16; m <<= 1) v = fmaxf(v, __shfl_xor(v, m, 32));
  return v;
}
__device__ __forceinline__ float hsum16(float v) {
#pragma unroll
  for (int m = 1; m < 16; m <<= 1) v += __shfl_xor(v, m, 32);
  return v;
}

#define TM 128
#define TN 128
#define TK 32

// Tensor Data Mover: async-copy one TM x TK bf16 tile (row-major, leading dim
// K elements) from global to LDS. 2D descriptor, data_size=2B. Issued by one
// wave; completion tracked with TENSORcnt.
__device__ __forceinline__ void tdm_load_bf16_tile(unsigned lds_off,
                                                   const unsigned short* gp,
                                                   int K, int Mtot) {
  unsigned long long ga = (unsigned long long)gp;
  v4u g0;
  g0.x = 1u;                                                 // count=1 (user D#)
  g0.y = lds_off;                                            // lds_addr [63:32]
  g0.z = (unsigned)ga;                                       // global_addr lo
  g0.w = ((unsigned)(ga >> 32) & 0x01FFFFFFu) | (2u << 30);  // addr hi, type=2
  v8i_ g1;
  g1[0] = 0x00010000;                                  // data_size=1 (2 bytes)
  g1[1] = (K & 0xFFFF) << 16;                          // tensor_dim0 lo16
  g1[2] = ((unsigned)K >> 16) | ((Mtot & 0xFFFF) << 16);  // dim0 hi, dim1 lo
  g1[3] = ((unsigned)Mtot >> 16) | (TK << 16);         // dim1 hi, tile_dim0=32
  g1[4] = TM;                                          // tile_dim1=128, dim2=0
  g1[5] = K;                                           // tensor_dim0_stride
  g1[6] = 0;
  g1[7] = 0;
  v4i_ z4 = {0, 0, 0, 0};
  v8i_ z8 = {0, 0, 0, 0, 0, 0, 0, 0};
  __builtin_amdgcn_tensor_load_to_lds(g0, g1, z4, z4, z8, 0);
}

// W-tile staging: 4x float4 per thread (128x32 f32 tile), convert->bf16 LDS.
__device__ __forceinline__ void stage_w_load(const float* __restrict__ W,
                                             int n0, int brow, int bcol, int K,
                                             int kk, float4 t4[4]) {
#pragma unroll
  for (int it = 0; it < 4; ++it)
    t4[it] = *(const float4*)&W[(size_t)(n0 + brow + 32 * it) * K + kk + bcol];
}
__device__ __forceinline__ void stage_w_store(unsigned short* bs, int brow,
                                              int bcol, const float4 t4[4]) {
#pragma unroll
  for (int it = 0; it < 4; ++it) {
    uint2 pk;
    pk.x = pack_bf2(t4[it].x, t4[it].y);
    pk.y = pack_bf2(t4[it].z, t4[it].w);
    *(uint2*)&bs[(brow + 32 * it) * TK + bcol] = pk;
  }
}

// ---------------------------------------------------------------- rmsnorm ---
__global__ __launch_bounds__(256) void rmsnorm_k(const float* __restrict__ x,
                                                 const float* __restrict__ w,
                                                 unsigned short* __restrict__ obf,
                                                 float* __restrict__ of,
                                                 int outIsBf) {
  int row = blockIdx.x;
  const float* xr = x + (size_t)row * HID_;
  __shared__ float red[256];
  int i0 = threadIdx.x * 8;  // 256 threads x 8 contiguous = 2048
  float4 a = *(const float4*)&xr[i0];
  float4 b = *(const float4*)&xr[i0 + 4];
  float ss = a.x * a.x + a.y * a.y + a.z * a.z + a.w * a.w +
             b.x * b.x + b.y * b.y + b.z * b.z + b.w * b.w;
  red[threadIdx.x] = ss;
  __syncthreads();
  for (int s = 128; s > 0; s >>= 1) {
    if ((int)threadIdx.x < s) red[threadIdx.x] += red[threadIdx.x + s];
    __syncthreads();
  }
  float rs = rsqrtf(red[0] * (1.0f / HID_) + 1e-6f);
  float4 wa = *(const float4*)&w[i0];
  float4 wb = *(const float4*)&w[i0 + 4];
  float o0 = a.x * rs * wa.x, o1 = a.y * rs * wa.y;
  float o2 = a.z * rs * wa.z, o3 = a.w * rs * wa.w;
  float o4 = b.x * rs * wb.x, o5 = b.y * rs * wb.y;
  float o6 = b.z * rs * wb.z, o7 = b.w * rs * wb.w;
  if (outIsBf) {
    uint4 q;
    q.x = pack_bf2(o0, o1);
    q.y = pack_bf2(o2, o3);
    q.z = pack_bf2(o4, o5);
    q.w = pack_bf2(o6, o7);
    *(uint4*)&obf[(size_t)row * HID_ + i0] = q;
  } else {
    float4 qa; qa.x = o0; qa.y = o1; qa.z = o2; qa.w = o3;
    float4 qb; qb.x = o4; qb.y = o5; qb.z = o6; qb.w = o7;
    *(float4*)&of[(size_t)row * HID_ + i0] = qa;
    *(float4*)&of[(size_t)row * HID_ + i0 + 4] = qb;
  }
}

// ------------------------------------------------------------------- GEMM ---
// C[M,N] = A[M,K](bf16) * W[N,K](f32)^T (+bias).
// mode 0: Cf = v;  mode 2: Cf += v (residual);
// mode 3: Cb = bf16(silu(Gin) * v) (fused SwiGLU up-projection).
// Double-buffered LDS pipeline: A tile via Tensor Data Mover (async DMA,
// TENSORcnt), W tile via vectorized global loads issued before the WMMAs of
// the previous tile so their latency hides under the matrix pipe.
__global__ __launch_bounds__(256) void gemm_bf16_k(
    const unsigned short* __restrict__ A, const float* __restrict__ W,
    const float* __restrict__ bias, float* __restrict__ Cf,
    unsigned short* __restrict__ Cb, const float* __restrict__ Gin,
    int M, int N, int K, int mode) {
  __shared__ unsigned short As[2][TM * TK];
  __shared__ unsigned short Bs[2][TN * TK];
  int tid = threadIdx.x;
  int lane = tid & 31, w = tid >> 5;
  int wm = w & 1, wn = w >> 1;  // 2 x 4 wave grid
  int m0 = blockIdx.y * TM, n0 = blockIdx.x * TN;
  unsigned as_off0 = (unsigned)(unsigned long long)(void*)&As[0][0];
  unsigned as_off1 = (unsigned)(unsigned long long)(void*)&As[1][0];

  floatx8 acc[4][2];
#pragma unroll
  for (int mi = 0; mi < 4; ++mi)
#pragma unroll
    for (int ni = 0; ni < 2; ++ni) acc[mi][ni] = floatx8{};

  int brow = tid >> 3;       // 0..31
  int bcol = (tid & 7) * 4;  // 0,4,...,28

  // ---- prologue: stage tile 0 into buffer 0
  {
    if (tid < 32) tdm_load_bf16_tile(as_off0, &A[(size_t)m0 * K], K, M);
    float4 t4[4];
    stage_w_load(W, n0, brow, bcol, K, 0, t4);
    stage_w_store(&Bs[0][0], brow, bcol, t4);
    if (tid < 32) __builtin_amdgcn_s_wait_tensorcnt(0);
    __syncthreads();
  }

  int nb = K / TK;
  for (int ib = 0; ib < nb; ++ib) {
    int cur = ib & 1;
    int kk2 = (ib + 1) * TK;
    bool have_next = (ib + 1 < nb);

    float4 t4[4];
    if (have_next) {
      stage_w_load(W, n0, brow, bcol, K, kk2, t4);  // in flight during WMMAs
      if (tid < 32)
        tdm_load_bf16_tile(cur ? as_off0 : as_off1,
                           &A[(size_t)m0 * K + kk2], K, M);
      if (kk2 + TK < K)
        __builtin_prefetch(&W[(size_t)(n0 + (tid >> 1)) * K + kk2 + TK], 0, 0);
    }

    // ---- compute on current buffers
    const unsigned short* as = &As[cur][0];
    const unsigned short* bs = &Bs[cur][0];
    bf16x16 af[4], bfg[2];
#pragma unroll
    for (int mi = 0; mi < 4; ++mi)
      af[mi] = load_frag(as + (wm * 64 + mi * 16) * TK, TK);
#pragma unroll
    for (int ni = 0; ni < 2; ++ni)
      bfg[ni] = load_frag(bs + (wn * 32 + ni * 16) * TK, TK);
#pragma unroll
    for (int mi = 0; mi < 4; ++mi)
#pragma unroll
      for (int ni = 0; ni < 2; ++ni)
        acc[mi][ni] = wmma_bf16(af[mi], bfg[ni], acc[mi][ni]);

    if (have_next) {
      stage_w_store(&Bs[cur ^ 1][0], brow, bcol, t4);
      if (tid < 32) __builtin_amdgcn_s_wait_tensorcnt(0);
    }
    __syncthreads();
  }

  int nl = lane & 15, half = lane >> 4;
#pragma unroll
  for (int mi = 0; mi < 4; ++mi)
#pragma unroll
    for (int ni = 0; ni < 2; ++ni) {
      int n = n0 + wn * 32 + ni * 16 + nl;
      float bv = bias ? bias[n] : 0.f;
#pragma unroll
      for (int r = 0; r < 8; ++r) {
        int m = m0 + wm * 64 + mi * 16 + r + 8 * half;
        float v = acc[mi][ni][r] + bv;
        size_t off = (size_t)m * N + n;
        if (mode == 0)      Cf[off] = v;
        else if (mode == 2) Cf[off] += v;
        else {  // mode 3: fused silu(gate) * up -> bf16
          float g = Gin[off];
          Cb[off] = f2bf((g / (1.f + __expf(-g))) * v);
        }
      }
    }
}

// ------------------------------------------------------------- RoPE / V^T ---
__global__ __launch_bounds__(256) void rope_k(const float* __restrict__ in,
                                              const float* __restrict__ cs,
                                              const float* __restrict__ sn,
                                              unsigned short* __restrict__ out,
                                              int nheads) {
  int idx = blockIdx.x * 256 + threadIdx.x;  // B*T*nheads*HD threads
  int d = idx % HD_;
  int h = (idx / HD_) % nheads;
  int t = (idx / (HD_ * nheads)) % T_;
  int b = idx / (HD_ * nheads * T_);
  size_t base = ((size_t)(b * T_ + t) * nheads + h) * HD_;
  float v = in[base + d];
  float rot = (d < HD_ / 2) ? -in[base + d + HD_ / 2] : in[base + d - HD_ / 2];
  size_t ct = (size_t)(b * T_ + t) * HD_ + d;
  out[((size_t)(b * nheads + h) * T_ + t) * HD_ + d] =
      f2bf(v * cs[ct] + rot * sn[ct]);
}

__global__ __launch_bounds__(256) void castv_k(const float* __restrict__ in,
                                               unsigned short* __restrict__ out) {
  int idx = blockIdx.x * 256 + threadIdx.x;  // B*T*NKV*HD threads
  int d = idx % HD_;
  int h = (idx / HD_) % NKV_;
  int t = (idx / (HD_ * NKV_)) % T_;
  int b = idx / (HD_ * NKV_ * T_);
  float v = in[((size_t)(b * T_ + t) * NKV_ + h) * HD_ + d];
  out[((size_t)(b * NKV_ + h) * HD_ + d) * T_ + t] = f2bf(v);  // [B,NKV,HD,T]
}

// -------------------------------------------------------- flash attention ---
// One wave per 16-row query tile x full HD. Online softmax; QK^T and P*V via
// bf16 WMMA. V pre-transposed to [HD,T] so PV B-operand fragments load
// row-major.
__global__ __launch_bounds__(32) void flash_attn_k(
    const unsigned short* __restrict__ Q,   // [B,NH,T,HD] bf16
    const unsigned short* __restrict__ Kc,  // [B,NKV,T,HD] bf16
    const unsigned short* __restrict__ Vt,  // [B,NKV,HD,T] bf16
    unsigned short* __restrict__ O) {       // [B,T,NH*HD] bf16
  __shared__ unsigned short Plds[16 * 32];
  int lane = threadIdx.x & 31;
  int nl = lane & 15, half = lane >> 4;
  int q0 = blockIdx.x * 16;
  int bh = blockIdx.y;
  int b = bh / NH_, h = bh % NH_, hk = h / (NH_ / NKV_);

  const unsigned short* qb = Q + ((size_t)(b * NH_ + h) * T_ + q0) * HD_;
  const unsigned short* kb = Kc + (size_t)(b * NKV_ + hk) * T_ * HD_;
  const unsigned short* vb = Vt + (size_t)(b * NKV_ + hk) * HD_ * T_;

  bf16x16 qf[4];
#pragma unroll
  for (int s = 0; s < 4; ++s) qf[s] = load_frag(qb + 32 * s, HD_);

  floatx8 o_[8] = {};
  float mrow[8], lrow[8];
#pragma unroll
  for (int r = 0; r < 8; ++r) { mrow[r] = -1e30f; lrow[r] = 0.f; }
  const float sc = 0.08838834764831845f;  // 1/sqrt(128)

  for (int k0 = 0; k0 <= q0; k0 += 32) {
    floatx8 s0{}, s1{};
#pragma unroll
    for (int s = 0; s < 4; ++s) {
      bf16x16 kf0 = load_frag(kb + (size_t)k0 * HD_ + 32 * s, HD_);
      bf16x16 kf1 = load_frag(kb + (size_t)(k0 + 16) * HD_ + 32 * s, HD_);
      s0 = wmma_bf16(qf[s], kf0, s0);
      s1 = wmma_bf16(qf[s], kf1, s1);
    }
    float p0[8], p1[8], corr[8];
#pragma unroll
    for (int r = 0; r < 8; ++r) {
      int m = q0 + r + 8 * half;
      float v0 = s0[r] * sc; if (k0 + nl > m)      v0 = -1e30f;
      float v1 = s1[r] * sc; if (k0 + 16 + nl > m) v1 = -1e30f;
      float rmax = hmax16(fmaxf(v0, v1));
      float mn = fmaxf(mrow[r], rmax);
      float cr = __expf(mrow[r] - mn);
      p0[r] = __expf(v0 - mn);
      p1[r] = __expf(v1 - mn);
      lrow[r] = lrow[r] * cr + hsum16(p0[r] + p1[r]);
      mrow[r] = mn;
      corr[r] = cr;
    }
#pragma unroll
    for (int i = 0; i < 8; ++i)
#pragma unroll
      for (int r = 0; r < 8; ++r) o_[i][r] *= corr[r];

    // stage P (C-layout) through LDS, reload in A-operand layout
#pragma unroll
    for (int r = 0; r < 8; ++r) {
      int row = r + 8 * half;
      Plds[row * 32 + nl]      = f2bf(p0[r]);
      Plds[row * 32 + 16 + nl] = f2bf(p1[r]);
    }
    __syncthreads();
    bf16x16 pf = load_frag(Plds, 32);
    __syncthreads();
#pragma unroll
    for (int i = 0; i < 8; ++i) {
      bf16x16 vf = load_frag(vb + (size_t)(16 * i) * T_ + k0, T_);
      o_[i] = wmma_bf16(pf, vf, o_[i]);
    }
  }

  float invl[8];
#pragma unroll
  for (int r = 0; r < 8; ++r) invl[r] = 1.0f / lrow[r];
  unsigned short* ob = O + ((size_t)(b * T_) + q0) * (NH_ * HD_) + h * HD_;
#pragma unroll
  for (int i = 0; i < 8; ++i)
#pragma unroll
    for (int r = 0; r < 8; ++r)
      ob[(size_t)(r + 8 * half) * (NH_ * HD_) + 16 * i + nl] =
          f2bf(o_[i][r] * invl[r]);
}

// ------------------------------------------------------------------ launch ---
extern "C" void kernel_launch(void* const* d_in, const int* in_sizes, int n_in,
                              void* d_out, int out_size, void* d_ws, size_t ws_size,
                              hipStream_t stream) {
  const float* hs    = (const float*)d_in[0];
  const float* cosb  = (const float*)d_in[1];
  const float* sinb  = (const float*)d_in[2];
  const float* qw    = (const float*)d_in[3];
  const float* qbias = (const float*)d_in[4];
  const float* kw    = (const float*)d_in[5];
  const float* kbias = (const float*)d_in[6];
  const float* vw    = (const float*)d_in[7];
  const float* vbias = (const float*)d_in[8];
  const float* ow    = (const float*)d_in[9];
  const float* gw    = (const float*)d_in[10];
  const float* uw    = (const float*)d_in[11];
  const float* dw    = (const float*)d_in[12];
  const float* ln1   = (const float*)d_in[13];
  const float* ln2   = (const float*)d_in[14];
  const float* normw = (const float*)d_in[15];
  float* out = (float*)d_out;

  char* p = (char*)d_ws;
  float* x            = (float*)p;          p += (size_t)BT_ * HID_ * 4;   // 16MB
  unsigned short* hn  = (unsigned short*)p; p += (size_t)BT_ * HID_ * 2;   //  8MB
  float* tmp          = (float*)p;          p += (size_t)BT_ * HID_ * 4;   // 16MB
  unsigned short* qbf = (unsigned short*)p; p += (size_t)BT_ * NH_ * HD_ * 2;   // 8MB
  unsigned short* kbf = (unsigned short*)p; p += (size_t)BT_ * NKV_ * HD_ * 2;  // 2MB
  unsigned short* vtb = (unsigned short*)p; p += (size_t)BT_ * NKV_ * HD_ * 2;  // 2MB
  unsigned short* att = (unsigned short*)p; p += (size_t)BT_ * NH_ * HD_ * 2;   // 8MB
  float* gbuf         = (float*)p;          p += (size_t)BT_ * INTER_ * 4;      // 46MB
  unsigned short* act = (unsigned short*)p; p += (size_t)BT_ * INTER_ * 2;      // 23MB

  (void)hipMemcpyAsync(x, hs, (size_t)BT_ * HID_ * 4, hipMemcpyDeviceToDevice,
                       stream);

  for (int l = 0; l < L_; ++l) {
    const float* qw_l = qw + (size_t)l * (NH_ * HD_) * HID_;
    const float* kw_l = kw + (size_t)l * (NKV_ * HD_) * HID_;
    const float* vw_l = vw + (size_t)l * (NKV_ * HD_) * HID_;
    const float* ow_l = ow + (size_t)l * HID_ * (NH_ * HD_);
    const float* gw_l = gw + (size_t)l * INTER_ * HID_;
    const float* uw_l = uw + (size_t)l * INTER_ * HID_;
    const float* dw_l = dw + (size_t)l * HID_ * INTER_;

    // attn block
    rmsnorm_k<<<BT_, 256, 0, stream>>>(x, ln1 + (size_t)l * HID_, hn, nullptr, 1);
    gemm_bf16_k<<<dim3((NH_ * HD_) / TN, BT_ / TM), 256, 0, stream>>>(
        hn, qw_l, qbias + (size_t)l * NH_ * HD_, tmp, nullptr, nullptr,
        BT_, NH_ * HD_, HID_, 0);
    rope_k<<<(BT_ * NH_ * HD_) / 256, 256, 0, stream>>>(tmp, cosb, sinb, qbf, NH_);
    gemm_bf16_k<<<dim3((NKV_ * HD_) / TN, BT_ / TM), 256, 0, stream>>>(
        hn, kw_l, kbias + (size_t)l * NKV_ * HD_, tmp, nullptr, nullptr,
        BT_, NKV_ * HD_, HID_, 0);
    rope_k<<<(BT_ * NKV_ * HD_) / 256, 256, 0, stream>>>(tmp, cosb, sinb, kbf, NKV_);
    gemm_bf16_k<<<dim3((NKV_ * HD_) / TN, BT_ / TM), 256, 0, stream>>>(
        hn, vw_l, vbias + (size_t)l * NKV_ * HD_, tmp, nullptr, nullptr,
        BT_, NKV_ * HD_, HID_, 0);
    castv_k<<<(BT_ * NKV_ * HD_) / 256, 256, 0, stream>>>(tmp, vtb);
    flash_attn_k<<<dim3(T_ / 16, B_ * NH_), 32, 0, stream>>>(qbf, kbf, vtb, att);
    gemm_bf16_k<<<dim3(HID_ / TN, BT_ / TM), 256, 0, stream>>>(
        att, ow_l, nullptr, x, nullptr, nullptr,
        BT_, HID_, NH_ * HD_, 2);  // x += attn @ ow^T

    // mlp block
    rmsnorm_k<<<BT_, 256, 0, stream>>>(x, ln2 + (size_t)l * HID_, hn, nullptr, 1);
    gemm_bf16_k<<<dim3(INTER_ / TN, BT_ / TM), 256, 0, stream>>>(
        hn, gw_l, nullptr, gbuf, nullptr, nullptr, BT_, INTER_, HID_, 0);
    gemm_bf16_k<<<dim3(INTER_ / TN, BT_ / TM), 256, 0, stream>>>(
        hn, uw_l, nullptr, nullptr, act, gbuf, BT_, INTER_, HID_, 3);  // fused
    gemm_bf16_k<<<dim3(HID_ / TN, BT_ / TM), 256, 0, stream>>>(
        act, dw_l, nullptr, x, nullptr, nullptr,
        BT_, HID_, INTER_, 2);     // x += act @ dw^T
  }

  rmsnorm_k<<<BT_, 256, 0, stream>>>(x, normw, nullptr, out, 0);
  (void)in_sizes; (void)n_in; (void)out_size; (void)ws_size;
}
